// NGPTAttention_65687229825548
// MI455X (gfx1250) — matrix operand order repaired and checked by reading, not compiled, not run on hardware
//
#include <hip/hip_runtime.h>
#include <math.h>

typedef __attribute__((ext_vector_type(16))) __bf16         v16bf;
typedef __attribute__((ext_vector_type(8)))  float          v8f;
typedef __attribute__((ext_vector_type(8)))  unsigned short ushort8;
typedef __attribute__((ext_vector_type(4)))  float          f32x4;

#define T_SEQ  2048
#define NHEAD  16
#define DMODEL 1024
#define LOG2E  1.44269504088896340736f

// ---------------------------------------------------------------- helpers

static __device__ __forceinline__ unsigned short bf_bits(float f) {
  union { __bf16 b; unsigned short u; } cv;
  cv.b = (__bf16)f;
  return cv.u;
}

static __device__ __forceinline__ v16bf pack_bf16(ushort8 lo, ushort8 hi) {
  union { v16bf v; ushort8 h[2]; } u;
  u.h[0] = lo; u.h[1] = hi;
  return u.v;
}

static __device__ __forceinline__ v8f wmma_bf16(v16bf a, v16bf b, v8f c) {
  // D(f32 16x16) = A(bf16 16x32) * B(bf16 32x16) + C
  return __builtin_amdgcn_wmma_f32_16x16x32_bf16(false, a, false, b,
                                                 (short)0, c, false, false);
}

// --------------------------------------------- kernel 0: fp32 -> bf16 pass
__global__ __launch_bounds__(256) void cvt_bf16_kernel(
    const float* __restrict__ src, unsigned short* __restrict__ dst, int n8) {
  const int i = blockIdx.x * blockDim.x + threadIdx.x;
  if (i >= n8) return;
  const f32x4* p = (const f32x4*)(src + (size_t)i * 8);
  f32x4 a = p[0], b = p[1];
  ushort8 o;
#pragma unroll
  for (int j = 0; j < 4; ++j) {
    o[j]     = bf_bits(a[j]);
    o[j + 4] = bf_bits(b[j]);
  }
  *(ushort8*)(dst + (size_t)i * 8) = o;
}

// ---------------------------------------------------------- kernel 1: QKV
// out[m][n] = sum_k x[m][k] * W[n][k]; all operands pre-converted bf16.
// Wave tile 32x64 (2x4 WMMA frags); block tile 64x256 (8 waves = 2x4).
// z<2: per-head L2 normalize, write bf16 (B,H,T,D). z==2: write V^T (B,H,D,T).
__global__ __launch_bounds__(256) void ngpt_qkv_kernel(
    const unsigned short* __restrict__ xb,
    const unsigned short* __restrict__ Wqb,
    const unsigned short* __restrict__ Wkb,
    const unsigned short* __restrict__ Wvb,
    unsigned short* __restrict__ qout, unsigned short* __restrict__ kout,
    unsigned short* __restrict__ vtout) {
  __shared__ float ctile[64][260];  // 64x256 tile + pad (~65 KB of 320 KB)
  const int z = blockIdx.z;
  const unsigned short* __restrict__ W = (z == 0) ? Wqb : (z == 1) ? Wkb : Wvb;
  const int tid = threadIdx.x;
  const int w = tid >> 5, lane = tid & 31, ln = lane & 15, hi = lane >> 4;
  const int wm = w >> 2, wn = w & 3;               // 2x4 waves of 32x64 tiles
  const int row0 = blockIdx.x * 64 + wm * 32;      // token rows
  const int col0 = blockIdx.y * 256 + wn * 64;     // output features

  v8f acc[2][4];
#pragma unroll
  for (int a0 = 0; a0 < 2; ++a0)
#pragma unroll
    for (int a1 = 0; a1 < 4; ++a1)
#pragma unroll
      for (int e = 0; e < 8; ++e) acc[a0][a1][e] = 0.f;

  for (int kk = 0; kk < DMODEL; kk += 32) {
    v16bf afrag[2], bfrag[4];
#pragma unroll
    for (int ms = 0; ms < 2; ++ms) {
      const unsigned short* pr =
          xb + (size_t)(row0 + ms * 16 + ln) * DMODEL + kk + 16 * hi;
      afrag[ms] = pack_bf16(*(const ushort8*)pr, *(const ushort8*)(pr + 8));
    }
#pragma unroll
    for (int ns = 0; ns < 4; ++ns) {
      const unsigned short* pw =
          W + (size_t)(col0 + ns * 16 + ln) * DMODEL + kk + 16 * hi;
      bfrag[ns] = pack_bf16(*(const ushort8*)pw, *(const ushort8*)(pw + 8));
    }
#pragma unroll
    for (int ms = 0; ms < 2; ++ms)
#pragma unroll
      for (int ns = 0; ns < 4; ++ns)
        acc[ms][ns] = wmma_bf16(afrag[ms], bfrag[ns], acc[ms][ns]);
  }

  // stage C tile to LDS (C layout: lane%16 = N, 8*lane half + vgpr = M row)
#pragma unroll
  for (int ms = 0; ms < 2; ++ms)
#pragma unroll
    for (int ns = 0; ns < 4; ++ns)
#pragma unroll
      for (int r = 0; r < 8; ++r)
        ctile[wm * 32 + ms * 16 + 8 * hi + r][wn * 64 + ns * 16 + ln] =
            acc[ms][ns][r];
  __syncthreads();

  {
    const int hl = tid >> 6;          // head within 256-col tile (0..3)
    const int row = tid & 63;
    const int grow = blockIdx.x * 64 + row;
    const int bi = grow >> 11;        // / T_SEQ
    const int t = grow & (T_SEQ - 1);
    const int h = blockIdx.y * 4 + hl;
    if (z < 2) {
      float ss = 0.f;
#pragma unroll
      for (int d = 0; d < 64; ++d) {
        float v = ctile[row][hl * 64 + d];
        ss += v * v;
      }
      const float inv = 1.0f / fmaxf(sqrtf(ss), 1e-12f);
      unsigned short* dst =
          (z == 0 ? qout : kout) + ((size_t)(bi * NHEAD + h) * T_SEQ + t) * 64;
#pragma unroll
      for (int d = 0; d < 64; ++d)
        dst[d] = bf_bits(ctile[row][hl * 64 + d] * inv);
    } else {
      unsigned short* dst = vtout + (size_t)(bi * NHEAD + h) * 64 * T_SEQ + t;
#pragma unroll
      for (int d = 0; d < 64; ++d)
        dst[(size_t)d * T_SEQ] = bf_bits(ctile[row][hl * 64 + d]);
    }
  }
}

// --------------------------------------------------- kernel 2: attention
// Per-wave flash attention: wave owns 16 query rows, full online softmax.
__global__ __launch_bounds__(256) void ngpt_attn_kernel(
    const unsigned short* __restrict__ qb,
    const unsigned short* __restrict__ kb,
    const unsigned short* __restrict__ vt,
    const float* __restrict__ sqk,
    unsigned short* __restrict__ att) {
  __shared__ unsigned short pshared[8][16 * 64];  // per-wave P transpose strip
  const int tid = threadIdx.x;
  const int w = tid >> 5, lane = tid & 31, ln = lane & 15, hi = lane >> 4;
  const int bh = blockIdx.y;
  const int h = bh & (NHEAD - 1);
  const int b = bh >> 4;
  const int q0 = blockIdx.x * 128 + w * 16;        // wave's 16 query rows
  const float scale = sqk[h] * 8.0f;               // * sqrt(HEAD_DIM)

  // Q A-fragments, straight from global (rows contiguous in d)
  const unsigned short* qrow = qb + ((size_t)bh * T_SEQ + q0 + ln) * 64;
  v16bf qa[2];
#pragma unroll
  for (int s = 0; s < 2; ++s) {
    ushort8 c0 = *(const ushort8*)(qrow + 32 * s + 16 * hi);
    ushort8 c1 = *(const ushort8*)(qrow + 32 * s + 16 * hi + 8);
    qa[s] = pack_bf16(c0, c1);
  }

  v8f o[4];
  float m[8], l[8];
#pragma unroll
  for (int n = 0; n < 4; ++n)
#pragma unroll
    for (int e = 0; e < 8; ++e) o[n][e] = 0.f;
#pragma unroll
  for (int r = 0; r < 8; ++r) { m[r] = -1e30f; l[r] = 0.f; }

  unsigned short* pbase = &pshared[w][0];
  const int q_hi = q0 + 15;

  for (int k0 = 0; k0 <= q_hi; k0 += 64) {   // causal trip count, per wave
    // ---- S = Q K^T (64 keys)
    v8f sacc[4];
#pragma unroll
    for (int t4 = 0; t4 < 4; ++t4) {
      const unsigned short* krow =
          kb + ((size_t)bh * T_SEQ + k0 + 16 * t4 + ln) * 64;
      v8f s_;
#pragma unroll
      for (int e = 0; e < 8; ++e) s_[e] = 0.f;
#pragma unroll
      for (int s = 0; s < 2; ++s) {
        ushort8 c0 = *(const ushort8*)(krow + 32 * s + 16 * hi);
        ushort8 c1 = *(const ushort8*)(krow + 32 * s + 16 * hi + 8);
        s_ = wmma_bf16(qa[s], pack_bf16(c0, c1), s_);
      }
      sacc[t4] = s_;
    }

    // ---- scale + causal mask + online softmax (rows live in lane halves)
#pragma unroll
    for (int r = 0; r < 8; ++r) {
      const int row = q0 + 8 * hi + r;
      float mx = -1e30f;
#pragma unroll
      for (int t4 = 0; t4 < 4; ++t4) {
        float v = sacc[t4][r] * scale;
        const int key = k0 + 16 * t4 + ln;
        if (key > row) v = -1e30f;
        sacc[t4][r] = v;
        mx = fmaxf(mx, v);
      }
      mx = fmaxf(mx, __shfl_xor(mx, 1));
      mx = fmaxf(mx, __shfl_xor(mx, 2));
      mx = fmaxf(mx, __shfl_xor(mx, 4));
      mx = fmaxf(mx, __shfl_xor(mx, 8));
      const float mn = fmaxf(m[r], mx);
      const float alpha = exp2f((m[r] - mn) * LOG2E);
      float rs = 0.f;
#pragma unroll
      for (int t4 = 0; t4 < 4; ++t4) {
        float p = exp2f((sacc[t4][r] - mn) * LOG2E);
        sacc[t4][r] = p;
        rs += p;
      }
      rs += __shfl_xor(rs, 1);
      rs += __shfl_xor(rs, 2);
      rs += __shfl_xor(rs, 4);
      rs += __shfl_xor(rs, 8);
      l[r] = l[r] * alpha + rs;
      m[r] = mn;
#pragma unroll
      for (int n = 0; n < 4; ++n) o[n][r] *= alpha;   // O layout matches rows
    }

    // ---- P: C-layout -> LDS row-major -> reload as A-fragments (per-wave strip)
#pragma unroll
    for (int t4 = 0; t4 < 4; ++t4)
#pragma unroll
      for (int r = 0; r < 8; ++r)
        pbase[(8 * hi + r) * 64 + 16 * t4 + ln] = bf_bits(sacc[t4][r]);
    asm volatile("s_wait_dscnt 0" ::: "memory");

    v16bf pa[2];
#pragma unroll
    for (int s = 0; s < 2; ++s) {
      ushort8 c0 = *(const ushort8*)(pbase + ln * 64 + 32 * s + 16 * hi);
      ushort8 c1 = *(const ushort8*)(pbase + ln * 64 + 32 * s + 16 * hi + 8);
      pa[s] = pack_bf16(c0, c1);
    }

    // ---- O += P * V  (V stored transposed: B-frag rows contiguous in keys)
#pragma unroll
    for (int n = 0; n < 4; ++n) {
      const unsigned short* vrow =
          vt + ((size_t)bh * 64 + 16 * n + ln) * T_SEQ + k0;
#pragma unroll
      for (int s = 0; s < 2; ++s) {
        ushort8 c0 = *(const ushort8*)(vrow + 32 * s + 16 * hi);
        ushort8 c1 = *(const ushort8*)(vrow + 32 * s + 16 * hi + 8);
        o[n] = wmma_bf16(pa[s], pack_bf16(c0, c1), o[n]);
      }
    }
  }

  // ---- normalize and store attention output as bf16 (B,T,H*D)
#pragma unroll
  for (int r = 0; r < 8; ++r) {
    const int t = q0 + 8 * hi + r;
    unsigned short* dst = att + ((size_t)b * T_SEQ + t) * DMODEL + h * 64;
    const float invl = 1.0f / l[r];
#pragma unroll
    for (int n = 0; n < 4; ++n) dst[16 * n + ln] = bf_bits(o[n][r] * invl);
  }
}

// ------------------------------------------------ kernel 3: out projection
// Wave tile 32x64, block 64x256; bf16 A (att) and bf16 B (pre-converted Wo).
__global__ __launch_bounds__(256) void ngpt_oproj_kernel(
    const unsigned short* __restrict__ att,
    const unsigned short* __restrict__ Wob,
    float* __restrict__ out) {
  const int tid = threadIdx.x;
  const int w = tid >> 5, lane = tid & 31, ln = lane & 15, hi = lane >> 4;
  const int wm = w >> 2, wn = w & 3;
  const int row0 = blockIdx.x * 64 + wm * 32;
  const int col0 = blockIdx.y * 256 + wn * 64;

  v8f acc[2][4];
#pragma unroll
  for (int a0 = 0; a0 < 2; ++a0)
#pragma unroll
    for (int a1 = 0; a1 < 4; ++a1)
#pragma unroll
      for (int e = 0; e < 8; ++e) acc[a0][a1][e] = 0.f;

  for (int kk = 0; kk < DMODEL; kk += 32) {
    v16bf afrag[2], bfrag[4];
#pragma unroll
    for (int ms = 0; ms < 2; ++ms) {
      const unsigned short* pr =
          att + (size_t)(row0 + ms * 16 + ln) * DMODEL + kk + 16 * hi;
      afrag[ms] = pack_bf16(*(const ushort8*)pr, *(const ushort8*)(pr + 8));
    }
#pragma unroll
    for (int ns = 0; ns < 4; ++ns) {
      const unsigned short* pw =
          Wob + (size_t)(col0 + ns * 16 + ln) * DMODEL + kk + 16 * hi;
      bfrag[ns] = pack_bf16(*(const ushort8*)pw, *(const ushort8*)(pw + 8));
    }
#pragma unroll
    for (int ms = 0; ms < 2; ++ms)
#pragma unroll
      for (int ns = 0; ns < 4; ++ns)
        acc[ms][ns] = wmma_bf16(afrag[ms], bfrag[ns], acc[ms][ns]);
  }

#pragma unroll
  for (int ms = 0; ms < 2; ++ms)
#pragma unroll
    for (int ns = 0; ns < 4; ++ns)
#pragma unroll
      for (int r = 0; r < 8; ++r)
        out[(size_t)(row0 + ms * 16 + 8 * hi + r) * DMODEL + col0 + ns * 16 + ln] =
            acc[ms][ns][r];
}

// ----------------------------------------------------------------- launch
extern "C" void kernel_launch(void* const* d_in, const int* in_sizes, int n_in,
                              void* d_out, int out_size, void* d_ws,
                              size_t ws_size, hipStream_t stream) {
  (void)in_sizes; (void)n_in; (void)out_size; (void)ws_size;
  const float* x   = (const float*)d_in[0];
  // d_in[1] = additive causal mask: applied analytically in-kernel
  const float* Wq  = (const float*)d_in[2];
  const float* Wk  = (const float*)d_in[3];
  const float* Wv  = (const float*)d_in[4];
  const float* Wo  = (const float*)d_in[5];
  const float* sqk = (const float*)d_in[6];

  const size_t act = (size_t)2 * NHEAD * T_SEQ * 64;   // 4Mi activations
  const size_t wel = (size_t)DMODEL * DMODEL;          // 1Mi weights
  unsigned short* qb  = (unsigned short*)d_ws;
  unsigned short* kbf = qb + act;
  unsigned short* vt  = kbf + act;
  unsigned short* att = vt + act;
  unsigned short* xb  = att + act;
  unsigned short* Wqb = xb + act;
  unsigned short* Wkb = Wqb + wel;
  unsigned short* Wvb = Wkb + wel;
  unsigned short* Wob = Wvb + wel;

  // 0) one-shot fp32 -> bf16 conversion (bandwidth-trivial; removes cvt
  //    VALU work and halves load bytes in every GEMM inner loop)
  cvt_bf16_kernel<<<(int)(act / 8 / 256), 256, 0, stream>>>(x, xb, (int)(act / 8));
  cvt_bf16_kernel<<<(int)(wel / 8 / 256), 256, 0, stream>>>(Wq, Wqb, (int)(wel / 8));
  cvt_bf16_kernel<<<(int)(wel / 8 / 256), 256, 0, stream>>>(Wk, Wkb, (int)(wel / 8));
  cvt_bf16_kernel<<<(int)(wel / 8 / 256), 256, 0, stream>>>(Wv, Wvb, (int)(wel / 8));
  cvt_bf16_kernel<<<(int)(wel / 8 / 256), 256, 0, stream>>>(Wo, Wob, (int)(wel / 8));

  // 1) fused QKV projection + per-head L2 norm (z = 0:q, 1:k, 2:v-transposed)
  ngpt_qkv_kernel<<<dim3(64, 4, 3), 256, 0, stream>>>(xb, Wqb, Wkb, Wvb,
                                                      qb, kbf, vt);
  // 2) per-wave flash attention over (B*H, T/128) tiles
  ngpt_attn_kernel<<<dim3(16, 32), 256, 0, stream>>>(qb, kbf, vt, sqk, att);
  // 3) output projection to fp32
  ngpt_oproj_kernel<<<dim3(64, 4), 256, 0, stream>>>(att, Wob, (float*)d_out);
}
